// Embedder_89661737271701
// MI455X (gfx1250) — matrix operand order repaired and checked by reading, not compiled, not run on hardware
//
#include <hip/hip_runtime.h>
#include <math.h>

typedef __attribute__((ext_vector_type(16))) __bf16 v16bf;
typedef __attribute__((ext_vector_type(8)))  __bf16 v8bf;
typedef __attribute__((ext_vector_type(8)))  float  v8f;
typedef __attribute__((ext_vector_type(4)))  unsigned int v4u;   // for NT stores

#define BATCH 512
#define TSTEPS 512
#define FIN 64
#define HID 128
#define G3 384   // 3*HID

__device__ __forceinline__ float fast_sigmoid(float v) {
    return __builtin_amdgcn_rcpf(1.0f + __expf(-v));   // v_exp + v_rcp
}
__device__ __forceinline__ float fast_tanh(float v) {
#if __has_builtin(__builtin_amdgcn_tanhf)
    return __builtin_amdgcn_tanhf(v);                  // v_tanh_f32
#else
    const float e = __expf(-2.0f * v);
    return (1.0f - e) * __builtin_amdgcn_rcpf(1.0f + e);
#endif
}

// ---------------------------------------------------------------------------
// Pack fp32 weight W[K,N] into bf16 WMMA B-fragment order (32B/lane chunks).
// ---------------------------------------------------------------------------
__global__ void pack_frag_kernel(const float* __restrict__ W,
                                 __bf16* __restrict__ P, int K, int N) {
    int idx = blockIdx.x * 256 + threadIdx.x;       // over K*N
    if (idx >= K * N) return;
    const int j    = idx & 15;
    const int lane = (idx >> 4) & 31;
    const int f    = idx >> 9;
    const int kch  = K >> 5;
    const int kc   = f % kch;
    const int nt   = f / kch;
    const int k    = kc * 32 + (j >> 3) * 16 + ((lane >> 4) & 1) * 8 + (j & 7);
    const int n    = nt * 16 + (lane & 15);
    P[idx] = (__bf16)W[(size_t)k * N + n];
}

// ---------------------------------------------------------------------------
// x [B,T,F] fp32 -> xt [T,B,F] bf16 (time-major), 4 elems/thread
// ---------------------------------------------------------------------------
__global__ void xpose_cvt_kernel(const float* __restrict__ x,
                                 __bf16* __restrict__ xt) {
    size_t i = ((size_t)blockIdx.x * 256 + threadIdx.x) * 4;   // over T*B*F
    const int f = (int)(i & (FIN - 1));
    size_t r = i >> 6;
    const int b = (int)(r & (BATCH - 1));
    const int t = (int)(r >> 9);
    const float4 v = *(const float4*)(x + ((size_t)b * TSTEPS + t) * FIN + f);
    union { __bf16 h[4]; uint2 u; } cv;
    cv.h[0] = (__bf16)v.x; cv.h[1] = (__bf16)v.y;
    cv.h[2] = (__bf16)v.z; cv.h[3] = (__bf16)v.w;
    *(uint2*)(xt + i) = cv.u;
}

// ---------------------------------------------------------------------------
// Input projection -> bf16 gx in scan-friendly transposed tile layout:
//   gxTb[(bt*384 + c)*16 + m]   (bt = global 16-row tile, c = 0..383)
// Blocks walk tiles in REVERSE order so the low-t tiles are written last
// and sit in L2 when the scan starts. 4 M-tiles/block, B-frags hoisted.
// ---------------------------------------------------------------------------
template <int DIN>
__global__ __launch_bounds__(256) void proj_gemm_kernel(
    const __bf16* __restrict__ act, const __bf16* __restrict__ Wxp,
    const float* __restrict__ bias, __bf16* __restrict__ gxTb) {
    constexpr int KCH = DIN / 32;
    constexpr int MT  = 4;
    __shared__ __bf16 sA[MT * 16 * DIN];
    const int tid  = threadIdx.x;
    const int lane = tid & 31;
    const int w    = tid >> 5;
    const int hi   = (lane >> 4) & 1;
    const int nl   = lane & 15;
    const int blk  = (int)gridDim.x - 1 - (int)blockIdx.x;   // reverse walk
    const size_t m0 = (size_t)blk * (16 * MT);

    constexpr int NV4 = (MT * 16 * DIN) / 8;
    const uint4* srcv = (const uint4*)(act + m0 * DIN);
    for (int i = tid; i < NV4; i += 256) ((uint4*)sA)[i] = srcv[i];

    v16bf Bf[3][KCH];
#pragma unroll
    for (int i = 0; i < 3; ++i)
#pragma unroll
        for (int kc = 0; kc < KCH; ++kc)
            Bf[i][kc] = *(const v16bf*)(
                Wxp + (size_t)(((w * 3 + i) * KCH + kc) * 32 + lane) * 16);
    float bv[3];
#pragma unroll
    for (int i = 0; i < 3; ++i) bv[i] = bias[(w * 3 + i) * 16 + nl];
    __syncthreads();

#pragma unroll
    for (int mt = 0; mt < MT; ++mt) {
        v8f acc[3] = {};
#pragma unroll
        for (int kc = 0; kc < KCH; ++kc) {
            v16bf a;
#pragma unroll
            for (int j = 0; j < 16; ++j)
                a[j] = sA[(mt * 16 + nl) * DIN + kc * 32 + (j >> 3) * 16 + hi * 8 + (j & 7)];
#pragma unroll
            for (int i = 0; i < 3; ++i)
                acc[i] = __builtin_amdgcn_wmma_f32_16x16x32_bf16(
                    false, a, false, Bf[i][kc], (short)0, acc[i], false, false);
        }
        const size_t bt = (size_t)blk * MT + mt;
#pragma unroll
        for (int i = 0; i < 3; ++i) {
            const int cc = (w * 3 + i) * 16 + nl;
            union { __bf16 h[8]; uint4 u; } o;
#pragma unroll
            for (int r = 0; r < 8; ++r) o.h[r] = (__bf16)(acc[i][r] + bv[i]);
            *(uint4*)(gxTb + (bt * G3 + cc) * 16 + hi * 8) = o.u;
        }
    }
}

// ---------------------------------------------------------------------------
// GRU scan over T. 32 blocks x 8 waves. Wh in VGPRs, h master in registers,
// bf16 h double-buffered in LDS (1 barrier/step). Gate inputs (bf16, mostly
// L2-resident) are software-pipelined one full step ahead.
// ---------------------------------------------------------------------------
__device__ __forceinline__ void load_gates(const __bf16* __restrict__ base,
                                           float* gz, float* gr, float* gn) {
    const v8bf z = *(const v8bf*)(base);
    const v8bf r = *(const v8bf*)(base + HID * 16);
    const v8bf n = *(const v8bf*)(base + 2 * HID * 16);
#pragma unroll
    for (int k = 0; k < 8; ++k) {
        gz[k] = (float)z[k]; gr[k] = (float)r[k]; gn[k] = (float)n[k];
    }
}

__global__ __launch_bounds__(256) void gru_scan_kernel(
    const __bf16* __restrict__ gxTb,  // packed transposed tiles (incl. bias)
    const __bf16* __restrict__ Whp,   // packed fragments: 24 tiles x 4 kc
    __bf16* __restrict__ hout) {      // [T*B, 128]
    __shared__ __bf16 h_b[2][16][HID + 8];   // 272B rows: 16B aligned
    const int tid  = threadIdx.x;
    const int lane = tid & 31;
    const int w    = tid >> 5;
    const int hi   = (lane >> 4) & 1;
    const int nl   = lane & 15;
    const int bt   = blockIdx.x;
    const int b0   = bt * 16;
    const int c    = w * 16 + nl;

    // prologue: issue gate loads for t=0 while we set up
    float gz[8], gr[8], gn[8];
    load_gates(gxTb + ((size_t)bt * G3 + c) * 16 + hi * 8, gz, gr, gn);

    for (int i = tid; i < 16 * HID; i += 256)
        h_b[0][i >> 7][i & (HID - 1)] = (__bf16)0.0f;

    v16bf Bf[3][4];
#pragma unroll
    for (int g = 0; g < 3; ++g)
#pragma unroll
        for (int kc = 0; kc < 4; ++kc)
            Bf[g][kc] = *(const v16bf*)(
                Whp + (size_t)(((g * 8 + w) * 4 + kc) * 32 + lane) * 16);

    float hreg[8];
#pragma unroll
    for (int r = 0; r < 8; ++r) hreg[r] = 0.0f;
    __syncthreads();

    for (int t = 0; t < TSTEPS; ++t) {
        const int cur = t & 1, nxt = cur ^ 1;

        // (1) stream out h of step t-1 (buffer `cur` complete; NT: don't
        //     let streaming h evict the gx working set from L2)
        if (t > 0) {
            __bf16* ho = hout + ((size_t)(t - 1) * BATCH + b0) * HID;
            const int row = tid >> 4, ch = tid & 15;
            __builtin_nontemporal_store(*(const v4u*)&h_b[cur][row][ch * 8],
                                        (v4u*)(ho + row * HID + ch * 8));
        }

        // (2) issue NEXT step's gate loads now (full step of latency hiding)
        const int tn = (t + 1 < TSTEPS) ? t + 1 : t;
        float nz[8], nr[8], nn[8];
        load_gates(gxTb + ((size_t)(tn * 32 + bt) * G3 + c) * 16 + hi * 8,
                   nz, nr, nn);
        // prefetch t+2's tile toward L2 (96 x 128B lines)
        const int tp = (t + 2 < TSTEPS) ? t + 2 : t;
        if (tid < 96)
            __builtin_prefetch(gxTb + (size_t)(tp * 32 + bt) * G3 * 16 + tid * 64, 0, 0);

        // (3) gh = h @ Wh
        v8f az = {}, ar = {}, an = {};
#pragma unroll
        for (int kc = 0; kc < 4; ++kc) {
            v16bf a;
#pragma unroll
            for (int j = 0; j < 16; ++j)
                a[j] = h_b[cur][nl][kc * 32 + (j >> 3) * 16 + hi * 8 + (j & 7)];
            az = __builtin_amdgcn_wmma_f32_16x16x32_bf16(false, a, false, Bf[0][kc],
                                                         (short)0, az, false, false);
            ar = __builtin_amdgcn_wmma_f32_16x16x32_bf16(false, a, false, Bf[1][kc],
                                                         (short)0, ar, false, false);
            an = __builtin_amdgcn_wmma_f32_16x16x32_bf16(false, a, false, Bf[2][kc],
                                                         (short)0, an, false, false);
        }

        // (4) lane-local gate math; h master stays in registers
#pragma unroll
        for (int r = 0; r < 8; ++r) {
            const float z  = fast_sigmoid(gz[r] + az[r]);
            const float rr = fast_sigmoid(gr[r] + ar[r]);
            const float n  = fast_tanh(gn[r] + rr * an[r]);
            hreg[r] = z * hreg[r] + (1.0f - z) * n;
            h_b[nxt][r + hi * 8][c] = (__bf16)hreg[r];
        }
        __syncthreads();   // h_b[nxt] visible; reads of cur done

        // (5) rotate pipelined stage registers
#pragma unroll
        for (int r = 0; r < 8; ++r) { gz[r] = nz[r]; gr[r] = nr[r]; gn[r] = nn[r]; }
    }

    // final h (t = T-1) lives in buffer (TSTEPS & 1)
    {
        const int fin = TSTEPS & 1;   // == 0 for T=512
        __bf16* ho = hout + ((size_t)(TSTEPS - 1) * BATCH + b0) * HID;
        const int row = tid >> 4, ch = tid & 15;
        __builtin_nontemporal_store(*(const v4u*)&h_b[fin][row][ch * 8],
                                    (v4u*)(ho + row * HID + ch * 8));
    }
}

// ---------------------------------------------------------------------------
// Dense head: out[b,t,:] = sigmoid(h[t,b,:] @ Wd + bd). 4 M-tiles per block.
// ---------------------------------------------------------------------------
__global__ __launch_bounds__(256) void dense_head_kernel(
    const __bf16* __restrict__ h,    // [T*B, 128]
    const __bf16* __restrict__ Wdp,  // packed fragments: 8 tiles x 4 kc
    const float* __restrict__ bd,    // [128]
    float* __restrict__ out) {       // [B, T, 128]
    constexpr int MT = 4;
    __shared__ __bf16 sA[MT * 16 * HID];
    const int tid  = threadIdx.x;
    const int lane = tid & 31;
    const int w    = tid >> 5;
    const int hi   = (lane >> 4) & 1;
    const int nl   = lane & 15;
    const size_t m0 = (size_t)blockIdx.x * (16 * MT);
    const int t    = (int)(m0 >> 9);          // 64 rows share one t
    const int bb0  = (int)(m0 & (BATCH - 1));

    const uint4* srcv = (const uint4*)(h + m0 * HID);
    for (int i = tid; i < (MT * 16 * HID) / 8; i += 256) ((uint4*)sA)[i] = srcv[i];

    v16bf Bf[4];
#pragma unroll
    for (int kc = 0; kc < 4; ++kc)
        Bf[kc] = *(const v16bf*)(Wdp + (size_t)((w * 4 + kc) * 32 + lane) * 16);
    const int cb = w * 16;
    const float bv = bd[cb + nl];
    __syncthreads();

#pragma unroll
    for (int mt = 0; mt < MT; ++mt) {
        v8f acc = {};
#pragma unroll
        for (int kc = 0; kc < 4; ++kc) {
            v16bf a;
#pragma unroll
            for (int j = 0; j < 16; ++j)
                a[j] = sA[(mt * 16 + nl) * HID + kc * 32 + (j >> 3) * 16 + hi * 8 + (j & 7)];
            acc = __builtin_amdgcn_wmma_f32_16x16x32_bf16(false, a, false, Bf[kc],
                                                          (short)0, acc, false, false);
        }
#pragma unroll
        for (int r = 0; r < 8; ++r) {
            const int b = bb0 + mt * 16 + r + hi * 8;
            out[(size_t)b * TSTEPS * HID + (size_t)t * HID + cb + nl] =
                fast_sigmoid(acc[r] + bv);
        }
    }
}

// ---------------------------------------------------------------------------
// Host orchestration
// ---------------------------------------------------------------------------
extern "C" void kernel_launch(void* const* d_in, const int* in_sizes, int n_in,
                              void* d_out, int out_size, void* d_ws, size_t ws_size,
                              hipStream_t stream) {
    const float* x   = (const float*)d_in[0];
    const float* Wx0 = (const float*)d_in[1];
    const float* Wxr = (const float*)d_in[2];
    const float* Wh  = (const float*)d_in[3];
    const float* b   = (const float*)d_in[4];
    const float* Wd  = (const float*)d_in[5];
    const float* bd  = (const float*)d_in[6];
    float* out = (float*)d_out;

    char* ws = (char*)d_ws;
    __bf16* wxp   = (__bf16*)(ws);                 // 122880 bf16
    __bf16* whp   = (__bf16*)(ws + 245760);        // 147456 bf16
    __bf16* wdp   = (__bf16*)(ws + 540672);        // 16384 bf16
    __bf16* xt_bf = (__bf16*)(ws + 573440);        // T*B*F bf16
    __bf16* hA    = (__bf16*)(ws + 34127872);      // T*B*H bf16
    __bf16* hB    = (__bf16*)(ws + 101236736);     // T*B*H bf16
    __bf16* gxTb  = (__bf16*)(ws + 168345600);     // T*B*384 bf16 (packed)

    const int L1OFF = FIN * G3;            // 24576
    const int LW    = HID * G3;            // 49152

    pack_frag_kernel<<<(FIN * G3 + 255) / 256, 256, 0, stream>>>(Wx0, wxp, FIN, G3);
    pack_frag_kernel<<<(LW + 255) / 256, 256, 0, stream>>>(Wxr, wxp + L1OFF, HID, G3);
    pack_frag_kernel<<<(LW + 255) / 256, 256, 0, stream>>>(Wxr + LW, wxp + L1OFF + LW, HID, G3);
    pack_frag_kernel<<<(LW + 255) / 256, 256, 0, stream>>>(Wh, whp, HID, G3);
    pack_frag_kernel<<<(LW + 255) / 256, 256, 0, stream>>>(Wh + LW, whp + LW, HID, G3);
    pack_frag_kernel<<<(LW + 255) / 256, 256, 0, stream>>>(Wh + 2 * LW, whp + 2 * LW, HID, G3);
    pack_frag_kernel<<<(HID * HID + 255) / 256, 256, 0, stream>>>(Wd, wdp, HID, HID);

    xpose_cvt_kernel<<<(TSTEPS * BATCH * FIN) / (256 * 4), 256, 0, stream>>>(x, xt_bf);

    const int mblocks = (TSTEPS * BATCH) / 64;     // 4096 (4 M-tiles/block)

    proj_gemm_kernel<FIN><<<mblocks, 256, 0, stream>>>(xt_bf, wxp, b, gxTb);
    gru_scan_kernel<<<BATCH / 16, 256, 0, stream>>>(gxTb, whp, hA);

    proj_gemm_kernel<HID><<<mblocks, 256, 0, stream>>>(hA, wxp + L1OFF, b + G3, gxTb);
    gru_scan_kernel<<<BATCH / 16, 256, 0, stream>>>(gxTb, whp + LW, hB);

    proj_gemm_kernel<HID><<<mblocks, 256, 0, stream>>>(hB, wxp + L1OFF + LW, b + 2 * G3, gxTb);
    gru_scan_kernel<<<BATCH / 16, 256, 0, stream>>>(gxTb, whp + 2 * LW, hA);

    dense_head_kernel<<<mblocks, 256, 0, stream>>>(hA, wdp, bd, out);
}
